// Linear_35175782154583
// MI455X (gfx1250) — matrix operand order repaired and checked by reading, not compiled, not run on hardware
//
#include <hip/hip_runtime.h>

// CDNA5 / gfx1250 wave32 WMMA GEMM:
//   out = w_scale * (input @ round(W)^T) + b_scale * round(bias)
// Fast path (ws_size permitting):
//   1) pre-pass: W f32 -> rint -> f16 into d_ws (exact: rounded weights are small ints)
//   2) GEMM: A staged f32->f16 in-loop (streamed once), B tiles copied with
//      GLOBAL_LOAD_ASYNC_TO_LDS_B128 (ASYNCcnt, double-buffered LDS),
//      v_wmma_f32_16x16x32_f16 compute, dequant+bias fused in epilogue.

typedef __attribute__((ext_vector_type(16))) _Float16 v16h;
typedef __attribute__((ext_vector_type(8)))  _Float16 v8h;
typedef __attribute__((ext_vector_type(4)))  _Float16 v4h;
typedef __attribute__((ext_vector_type(8)))  float    v8f;
typedef __attribute__((ext_vector_type(4)))  float    v4f;

#define TILE_M 128
#define TILE_N 128
#define TILE_K 32
#define LDSW   40   // halves per LDS row: 32 data + 8 pad -> 80B pitch, conflict-free b128

// ---------- CDNA5 async global->LDS copy (ASYNCcnt path) ----------
static __device__ __forceinline__
void async_copy_b128(unsigned lds_byte_addr, const void* sbase, unsigned voff_bytes) {
    // GVS form: mem = SADDR64 + VADDR32 ; LDS dest address in VDST vgpr
    asm volatile("global_load_async_to_lds_b128 %0, %1, %2"
                 :
                 : "v"(lds_byte_addr), "v"(voff_bytes), "s"(sbase)
                 : "memory");
}
template <int N>
static __device__ __forceinline__ void wait_asynccnt() {
    asm volatile("s_wait_asynccnt %0" :: "n"(N) : "memory");
}

// ---------- f32 -> f16 converters ----------
static __device__ __forceinline__ v8h cvt8(v4f x, v4f y) {
    v8h r;
    r[0] = (_Float16)x[0]; r[1] = (_Float16)x[1];
    r[2] = (_Float16)x[2]; r[3] = (_Float16)x[3];
    r[4] = (_Float16)y[0]; r[5] = (_Float16)y[1];
    r[6] = (_Float16)y[2]; r[7] = (_Float16)y[3];
    return r;
}
static __device__ __forceinline__ v8h cvt8_round(v4f x, v4f y) {
    v8h r;
    r[0] = (_Float16)__builtin_rintf(x[0]); r[1] = (_Float16)__builtin_rintf(x[1]);
    r[2] = (_Float16)__builtin_rintf(x[2]); r[3] = (_Float16)__builtin_rintf(x[3]);
    r[4] = (_Float16)__builtin_rintf(y[0]); r[5] = (_Float16)__builtin_rintf(y[1]);
    r[6] = (_Float16)__builtin_rintf(y[2]); r[7] = (_Float16)__builtin_rintf(y[3]);
    return r;
}

// ================= pre-pass: W2[i] = (f16) rint(W[i]) =================
extern "C" __global__ void __launch_bounds__(256)
quantize_w_f16(const float* __restrict__ W, _Float16* __restrict__ W2, int n4)
{
    int i = blockIdx.x * blockDim.x + threadIdx.x;
    if (i < n4) {
        v4f w = ((const v4f*)W)[i];
        v4h h;
        h[0] = (_Float16)__builtin_rintf(w[0]);
        h[1] = (_Float16)__builtin_rintf(w[1]);
        h[2] = (_Float16)__builtin_rintf(w[2]);
        h[3] = (_Float16)__builtin_rintf(w[3]);
        ((v4h*)W2)[i] = h;
    }
}

// ================= fast GEMM: f16 weights via async-to-LDS =================
extern "C" __global__ void __launch_bounds__(256)
linear_q_wmma_async(const float*    __restrict__ A,    // [M,K] f32
                    const _Float16* __restrict__ W2,   // [N,K] f16 (pre-rounded)
                    const float*    __restrict__ bias, // [N]   f32
                    const float*    __restrict__ wscale_p,
                    const float*    __restrict__ bscale_p,
                    float*          __restrict__ C,    // [M,N]
                    int M, int N, int K)
{
    __shared__ _Float16 lA[TILE_M * LDSW];
    __shared__ _Float16 lB[2 * TILE_N * LDSW];     // double-buffered async target

    const int tid  = threadIdx.x;
    const int lane = tid & 31;
    const int wave = tid >> 5;     // 8 wave32
    const int wm   = wave >> 2;    // 0..1 -> 64-row strip
    const int wn   = wave & 3;     // 0..3 -> 32-col strip

    const int bm = blockIdx.y * TILE_M;
    const int bn = blockIdx.x * TILE_N;

    const float ws = wscale_p[0];
    const float bs = bscale_p[0];

    // ---- A staging (f32 global -> regs -> f16 LDS): 2 threads/row, 16 f32 each
    const int srow  = tid >> 1;
    const int skoff = (tid & 1) << 4;
    const float* aP = A + (size_t)(bm + srow) * (size_t)K + skoff;
    v4f aR[4];
#pragma unroll
    for (int i = 0; i < 4; ++i) aR[i] = ((const v4f*)aP)[i];

    // ---- B async copy plan: tile = 128 rows x 64B; 512 b128-chunks; 2 per thread
    const unsigned K2    = (unsigned)K * 2u;             // row pitch in bytes (f16)
    const unsigned r0    = (unsigned)tid >> 2;           // rows 0..63
    const unsigned q0    = ((unsigned)tid & 3u) * 16u;   // 16B chunk in row
    const unsigned voff0 = r0 * K2 + q0;
    const unsigned voff1 = voff0 + 64u * K2;             // rows 64..127
    const unsigned loff0 = r0 * (LDSW * 2u) + q0;
    const unsigned loff1 = loff0 + 64u * (LDSW * 2u);
    const unsigned ldsB  = (unsigned)(uintptr_t)&lB[0];
    const unsigned bufBytes = TILE_N * LDSW * 2u;

    const _Float16* wTile = W2 + (size_t)bn * (size_t)K; // stage k0 adds k0 halves

    // prologue: async stage 0 -> buffer 0
    async_copy_b128(ldsB + loff0, wTile, voff0);
    async_copy_b128(ldsB + loff1, wTile, voff1);

    v8f acc[4][2];
#pragma unroll
    for (int i = 0; i < 4; ++i)
#pragma unroll
        for (int j = 0; j < 2; ++j)
            acc[i][j] = (v8f){0.f, 0.f, 0.f, 0.f, 0.f, 0.f, 0.f, 0.f};

    int buf = 0;
    for (int k0 = 0; k0 < K; k0 += TILE_K) {
        // stage A tile (convert f32 -> f16)
        *(v8h*)&lA[srow * LDSW + skoff]     = cvt8(aR[0], aR[1]);
        *(v8h*)&lA[srow * LDSW + skoff + 8] = cvt8(aR[2], aR[3]);

        if (k0 + TILE_K < K) {
            // issue next B stage into the other buffer (safe: last read 1 stage ago,
            // all waves passed the trailing barrier of the previous iteration)
            const _Float16* wNext = wTile + (k0 + TILE_K);
            const unsigned nxt = ldsB + (unsigned)(buf ^ 1) * bufBytes;
            async_copy_b128(nxt + loff0, wNext, voff0);
            async_copy_b128(nxt + loff1, wNext, voff1);
            wait_asynccnt<2>();      // current stage's 2 copies retired (in-order)
            // refill A regs for next stage
            aP += TILE_K;
#pragma unroll
            for (int i = 0; i < 4; ++i) aR[i] = ((const v4f*)aP)[i];
            __builtin_prefetch(aP + TILE_K, 0, 0);
        } else {
            wait_asynccnt<0>();
        }
        __syncthreads();

        // ---- fragments ----
        const _Float16* lb = lB + (size_t)buf * (TILE_N * LDSW);
        v16h af[4];
        {
            const int r16 = lane & 15;
            const int ak  = (lane >> 4) * 8;
#pragma unroll
            for (int mt = 0; mt < 4; ++mt) {
                const int row = wm * 64 + mt * 16 + r16;
                v8h lo = *(const v8h*)&lA[row * LDSW + ak];
                v8h hi = *(const v8h*)&lA[row * LDSW + ak + 16];
                af[mt] = __builtin_shufflevector(lo, hi, 0,1,2,3,4,5,6,7,8,9,10,11,12,13,14,15);
            }
        }
        v16h bf[2];
        {
            const int c16 = lane & 15;
            const int bk  = (lane >> 4) * 8;
#pragma unroll
            for (int nt = 0; nt < 2; ++nt) {
                const int col = wn * 32 + nt * 16 + c16;
                v8h lo = *(const v8h*)&lb[col * LDSW + bk];
                v8h hi = *(const v8h*)&lb[col * LDSW + bk + 16];
                bf[nt] = __builtin_shufflevector(lo, hi, 0,1,2,3,4,5,6,7,8,9,10,11,12,13,14,15);
            }
        }

#pragma unroll
        for (int mt = 0; mt < 4; ++mt)
#pragma unroll
            for (int nt = 0; nt < 2; ++nt)
                acc[mt][nt] = __builtin_amdgcn_wmma_f32_16x16x32_f16(
                    false, af[mt], false, bf[nt], (short)0, acc[mt][nt], false, false);

        __syncthreads();
        buf ^= 1;
    }

    // ---- epilogue: D layout row = v + 8*(lane>>4), col = lane&15 per 16x16 tile ----
#pragma unroll
    for (int nt = 0; nt < 2; ++nt) {
        const int col = bn + wn * 32 + nt * 16 + (lane & 15);
        const float bv = bs * __builtin_rintf(bias[col]);
#pragma unroll
        for (int mt = 0; mt < 4; ++mt) {
            const int rbase = bm + wm * 64 + mt * 16 + ((lane >> 4) << 3);
#pragma unroll
            for (int v = 0; v < 8; ++v)
                C[(size_t)(rbase + v) * (size_t)N + col] = ws * acc[mt][nt][v] + bv;
        }
    }
}

// ================= fallback (round-1 kernel): in-loop weight rounding =================
extern "C" __global__ void __launch_bounds__(256)
linear_q_wmma_fb(const float* __restrict__ A, const float* __restrict__ W,
                 const float* __restrict__ bias, const float* __restrict__ wscale_p,
                 const float* __restrict__ bscale_p, float* __restrict__ C,
                 int M, int N, int K)
{
    __shared__ _Float16 lA[TILE_M * LDSW];
    __shared__ _Float16 lB[TILE_N * LDSW];

    const int tid  = threadIdx.x;
    const int lane = tid & 31;
    const int wave = tid >> 5;
    const int wm   = wave >> 2;
    const int wn   = wave & 3;

    const int bm = blockIdx.y * TILE_M;
    const int bn = blockIdx.x * TILE_N;

    const int srow  = tid >> 1;
    const int skoff = (tid & 1) << 4;

    const float ws = wscale_p[0];
    const float bs = bscale_p[0];

    const float* aP = A + (size_t)(bm + srow) * (size_t)K + skoff;
    const float* wP = W + (size_t)(bn + srow) * (size_t)K + skoff;

    v4f aR[4], wR[4];
#pragma unroll
    for (int i = 0; i < 4; ++i) { aR[i] = ((const v4f*)aP)[i]; wR[i] = ((const v4f*)wP)[i]; }

    v8f acc[4][2];
#pragma unroll
    for (int i = 0; i < 4; ++i)
#pragma unroll
        for (int j = 0; j < 2; ++j)
            acc[i][j] = (v8f){0.f, 0.f, 0.f, 0.f, 0.f, 0.f, 0.f, 0.f};

    for (int k0 = 0; k0 < K; k0 += TILE_K) {
        *(v8h*)&lA[srow * LDSW + skoff]     = cvt8(aR[0], aR[1]);
        *(v8h*)&lA[srow * LDSW + skoff + 8] = cvt8(aR[2], aR[3]);
        *(v8h*)&lB[srow * LDSW + skoff]     = cvt8_round(wR[0], wR[1]);
        *(v8h*)&lB[srow * LDSW + skoff + 8] = cvt8_round(wR[2], wR[3]);
        __syncthreads();

        if (k0 + TILE_K < K) {
            aP += TILE_K; wP += TILE_K;
#pragma unroll
            for (int i = 0; i < 4; ++i) { aR[i] = ((const v4f*)aP)[i]; wR[i] = ((const v4f*)wP)[i]; }
            __builtin_prefetch(aP + TILE_K, 0, 0);
            __builtin_prefetch(wP + TILE_K, 0, 0);
        }

        v16h af[4];
        {
            const int r16 = lane & 15;
            const int ak  = (lane >> 4) * 8;
#pragma unroll
            for (int mt = 0; mt < 4; ++mt) {
                const int row = wm * 64 + mt * 16 + r16;
                v8h lo = *(const v8h*)&lA[row * LDSW + ak];
                v8h hi = *(const v8h*)&lA[row * LDSW + ak + 16];
                af[mt] = __builtin_shufflevector(lo, hi, 0,1,2,3,4,5,6,7,8,9,10,11,12,13,14,15);
            }
        }
        v16h bf[2];
        {
            const int c16 = lane & 15;
            const int bk  = (lane >> 4) * 8;
#pragma unroll
            for (int nt = 0; nt < 2; ++nt) {
                const int col = wn * 32 + nt * 16 + c16;
                v8h lo = *(const v8h*)&lB[col * LDSW + bk];
                v8h hi = *(const v8h*)&lB[col * LDSW + bk + 16];
                bf[nt] = __builtin_shufflevector(lo, hi, 0,1,2,3,4,5,6,7,8,9,10,11,12,13,14,15);
            }
        }

#pragma unroll
        for (int mt = 0; mt < 4; ++mt)
#pragma unroll
            for (int nt = 0; nt < 2; ++nt)
                acc[mt][nt] = __builtin_amdgcn_wmma_f32_16x16x32_f16(
                    false, af[mt], false, bf[nt], (short)0, acc[mt][nt], false, false);

        __syncthreads();
    }

#pragma unroll
    for (int nt = 0; nt < 2; ++nt) {
        const int col = bn + wn * 32 + nt * 16 + (lane & 15);
        const float bv = bs * __builtin_rintf(bias[col]);
#pragma unroll
        for (int mt = 0; mt < 4; ++mt) {
            const int rbase = bm + wm * 64 + mt * 16 + ((lane >> 4) << 3);
#pragma unroll
            for (int v = 0; v < 8; ++v)
                C[(size_t)(rbase + v) * (size_t)N + col] = ws * acc[mt][nt][v] + bv;
        }
    }
}

extern "C" void kernel_launch(void* const* d_in, const int* in_sizes, int n_in,
                              void* d_out, int out_size, void* d_ws, size_t ws_size,
                              hipStream_t stream) {
    const float* input  = (const float*)d_in[0];   // [M,K] f32
    const float* weight = (const float*)d_in[1];   // [N,K] f32
    const float* bias   = (const float*)d_in[2];   // [N]   f32
    const float* wscale = (const float*)d_in[3];   // [1]
    const float* bscale = (const float*)d_in[4];   // [1]
    float* out = (float*)d_out;

    const int N = in_sizes[2];              // out_f = 4096
    const int K = in_sizes[1] / N;          // in_f  = 4096
    const int M = in_sizes[0] / K;          // n_tok = 8192

    dim3 grid(N / TILE_N, M / TILE_M);
    const size_t wbytes = (size_t)N * (size_t)K * 2u;

    if (ws_size >= wbytes) {
        _Float16* W2 = (_Float16*)d_ws;
        const int n4 = (N * K) / 4;
        quantize_w_f16<<<dim3((n4 + 255) / 256), dim3(256), 0, stream>>>(weight, W2, n4);
        linear_q_wmma_async<<<grid, dim3(256), 0, stream>>>(input, W2, bias, wscale, bscale,
                                                            out, M, N, K);
    } else {
        linear_q_wmma_fb<<<grid, dim3(256), 0, stream>>>(input, weight, bias, wscale, bscale,
                                                         out, M, N, K);
    }
}
